// RoPEMultiheadAttention_16587163697375
// MI455X (gfx1250) — matrix operand order repaired and checked by reading, not compiled, not run on hardware
//
#include <hip/hip_runtime.h>
#include <hip/hip_fp16.h>

typedef __attribute__((ext_vector_type(16))) _Float16 v16h;
typedef __attribute__((ext_vector_type(8)))  _Float16 v8h;
typedef __attribute__((ext_vector_type(8)))  float    v8f;

#define WMMA_F16(a, b, c) \
    __builtin_amdgcn_wmma_f32_16x16x32_f16(false, (a), false, (b), (short)0, (c), false, false)

static constexpr int Bsz  = 4;
static constexpr int T    = 2048;
static constexpr int D    = 1024;
static constexpr int H    = 16;
static constexpr int HD   = 64;

#if defined(__gfx1250__) && __has_builtin(__builtin_amdgcn_tensor_load_to_lds) && \
    __has_builtin(__builtin_amdgcn_s_wait_tensorcnt)
#define USE_TDM 1
#else
#define USE_TDM 0
#endif

__device__ inline v8f zero8() {
    v8f z;
#pragma unroll
    for (int i = 0; i < 8; ++i) z[i] = 0.0f;
    return z;
}

// A-fragment (16x32 f16): lane m = lane&15 ; element j -> k = j + 8*(j>=8) + 8*(lane>=16)
__device__ inline v16h frag_a_from(const _Float16* base, int row_stride, int lane) {
    int m    = lane & 15;
    int koff = (lane & 16) >> 1;   // 0 or 8
    const _Float16* p = base + m * row_stride + koff;
    v8h lo = *(const v8h*)(p);
    v8h hi = *(const v8h*)(p + 16);
    v16h r;
#pragma unroll
    for (int j = 0; j < 8; ++j) { r[j] = lo[j]; r[j + 8] = hi[j]; }
    return r;
}

// B-fragment (32x16 f16): lane n = lane&15 ; element j -> k = j + 8*(j>=8) + 8*(lane>=16)
__device__ inline v16h frag_b_from(const _Float16* base, int row_stride, int lane) {
    int n    = lane & 15;
    int koff = (lane & 16) >> 1;   // 0 or 8
    const _Float16* p = base + n * row_stride + koff;
    v8h lo = *(const v8h*)(p);
    v8h hi = *(const v8h*)(p + 16);
    v16h r;
#pragma unroll
    for (int j = 0; j < 8; ++j) { r[j] = lo[j]; r[j + 8] = hi[j]; }
    return r;
}

#if USE_TDM
typedef __attribute__((ext_vector_type(4))) unsigned int u32x4;
typedef __attribute__((ext_vector_type(8))) int          i32x8;
typedef __attribute__((ext_vector_type(4))) int          i32x4;

// 2D f16 tile DMA: tile_k contiguous elements per row, tile_rows rows, row stride = stride elems.
// D# per CDNA5 ISA ch.8: group0 {count|lds_addr|global_addr|type=2}, group1 {dsize, dims, tile, strides}.
__device__ inline void tdm_load_tile(const _Float16* g, unsigned lds_byte_off,
                                     int tile_k, int tile_rows,
                                     int row_len, int total_rows, int stride) {
    unsigned long long ga = (unsigned long long)g;
    u32x4 g0 = { 1u,                               // count=1, user descriptor
                 lds_byte_off,                     // lds_addr (bytes)
                 (unsigned)ga,                     // global_addr[31:0]
                 (unsigned)(((ga >> 32) & 0x01FFFFFFull) | (2u << 30)) }; // addr[56:32] | type=2
    i32x8 g1;
    g1[0] = 0x00010000;                                                    // data_size=1 (2B), mask=0
    g1[1] = (int)((row_len & 0xFFFF) << 16);                               // tensor_dim0[15:0]
    g1[2] = (int)(((row_len >> 16) & 0xFFFF) | ((total_rows & 0xFFFF) << 16)); // dim0 hi | dim1 lo
    g1[3] = (int)(((total_rows >> 16) & 0xFFFF) | ((tile_k & 0xFFFF) << 16));  // dim1 hi | tile_dim0
    g1[4] = (int)(tile_rows & 0xFFFF);                                     // tile_dim1 (tile_dim2=0)
    g1[5] = (int)stride;                                                   // tensor_dim0_stride lo
    g1[6] = 0;                                                             // stride hi | dim1_stride lo
    g1[7] = 0;
    i32x4 z4 = {0, 0, 0, 0};
    i32x8 z8;
#pragma unroll
    for (int i = 0; i < 8; ++i) z8[i] = 0;
    // 6-arg toolchain variant: (group0, group1, group2, group3, group4, cpol)
    __builtin_amdgcn_tensor_load_to_lds(g0, g1, z4, z4, z8, 0);
}
__device__ inline unsigned lds_off_of(const void* p) {
    return (unsigned)(unsigned long long)p;        // generic LDS ptr: low 32 bits = LDS offset
}
#endif

__global__ void f32_to_f16(const float* __restrict__ in, _Float16* __restrict__ out, int n) {
    int i = blockIdx.x * blockDim.x + threadIdx.x;
    if (i < n) out[i] = (_Float16)in[i];
}

// ===========================================================================
// Shared WMMA-GEMM core: C(128x64) block tile, 4 waves, each wave 32(M)x64(N).
// Double-buffered LDS tiles, TDM DMA (wave 0) when available.
// A: [Mtot,1024] f16 row-major ; W: [Ntot,1024] f16 row-major ; C = A.W^T
// ===========================================================================
template <typename EPILOG>
__device__ inline void gemm_core(const _Float16* __restrict__ A,
                                 const _Float16* __restrict__ W,
                                 int Mtot, int Ntot, EPILOG epilog) {
    constexpr int K = D;
    __shared__ __align__(16) _Float16 As[2][128 * 32];
    __shared__ __align__(16) _Float16 Bs[2][64 * 32];

    const int tid  = threadIdx.x;
    const int lane = tid & 31;
    const int wave = tid >> 5;
    const int m0   = blockIdx.y * 128;
    const int n0   = blockIdx.x * 64;

    v8f c[2][4];
#pragma unroll
    for (int i = 0; i < 2; ++i)
#pragma unroll
        for (int j = 0; j < 4; ++j) c[i][j] = zero8();

#if USE_TDM
    if (wave == 0) {
        tdm_load_tile(&A[(size_t)m0 * K], lds_off_of(&As[0][0]), 32, 128, K, Mtot, K);
        tdm_load_tile(&W[(size_t)n0 * K], lds_off_of(&Bs[0][0]), 32, 64,  K, Ntot, K);
    }
#endif

    for (int k0 = 0; k0 < K; k0 += 32) {
        const int buf = (k0 >> 5) & 1;
#if USE_TDM
        if (wave == 0) {
            if (k0 + 32 < K) {
                tdm_load_tile(&A[(size_t)m0 * K + k0 + 32], lds_off_of(&As[buf ^ 1][0]),
                              32, 128, K, Mtot, K);
                tdm_load_tile(&W[(size_t)n0 * K + k0 + 32], lds_off_of(&Bs[buf ^ 1][0]),
                              32, 64, K, Ntot, K);
                __builtin_amdgcn_s_wait_tensorcnt(2);   // current buffer's pair done (in-order)
            } else {
                __builtin_amdgcn_s_wait_tensorcnt(0);
            }
        }
        __syncthreads();
#else
        __syncthreads();
#pragma unroll
        for (int it = 0; it < 4; ++it) {                // A: 4096 halves / 128 thr
            int cidx = tid + it * 128;
            int row  = cidx >> 2;
            int ch   = (cidx & 3) * 8;
            *(v8h*)&As[buf][row * 32 + ch] = *(const v8h*)&A[(size_t)(m0 + row) * K + k0 + ch];
        }
#pragma unroll
        for (int it = 0; it < 2; ++it) {                // B: 2048 halves / 128 thr
            int cidx = tid + it * 128;
            int row  = cidx >> 2;
            int ch   = (cidx & 3) * 8;
            *(v8h*)&Bs[buf][row * 32 + ch] = *(const v8h*)&W[(size_t)(n0 + row) * K + k0 + ch];
        }
        __syncthreads();
#endif
        v16h a0 = frag_a_from(&As[buf][(wave * 32) * 32],      32, lane);
        v16h a1 = frag_a_from(&As[buf][(wave * 32 + 16) * 32], 32, lane);
#pragma unroll
        for (int nt = 0; nt < 4; ++nt) {
            v16h b    = frag_b_from(&Bs[buf][nt * 16 * 32], 32, lane);
            c[0][nt]  = WMMA_F16(a0, b, c[0][nt]);
            c[1][nt]  = WMMA_F16(a1, b, c[1][nt]);
        }
#if USE_TDM
        __syncthreads();                                // readers done before next DMA overwrites
#endif
    }
    epilog(c, m0, n0, wave, lane);
}

// ---------------------------------------------------------------------------
// Kernel 1: qkv = x @ Wqkv^T + bqkv ; RoPE(q,k) ; scatter to Q/K/V [B,H,T,64] f16
// grid = (3072/64, 8192/128)
// ---------------------------------------------------------------------------
__global__ __launch_bounds__(128) void qkv_rope_kernel(
    const _Float16* __restrict__ X, const _Float16* __restrict__ W,
    const float* __restrict__ bias,
    _Float16* __restrict__ Qo, _Float16* __restrict__ Ko, _Float16* __restrict__ Vo)
{
    gemm_core(X, W, Bsz * T, 3 * D,
      [=] __device__ (v8f (&c)[2][4], int m0, int n0, int wave, int lane) {
        const int ncol  = lane & 15;
        const int half8 = (lane & 16) >> 1;
        const int which = n0 / D;                  // 0=q 1=k 2=v (64-col tile == one head)
        const int h     = (n0 % D) / HD;
        const float b0  = bias[n0 + ncol];
        const float b1  = bias[n0 + 16 + ncol];
        const float b2  = bias[n0 + 32 + ncol];
        const float b3  = bias[n0 + 48 + ncol];
#pragma unroll
        for (int rg = 0; rg < 2; ++rg) {
#pragma unroll
            for (int r = 0; r < 8; ++r) {
                int m     = m0 + wave * 32 + rg * 16 + r + half8;
                int b_idx = m >> 11;
                int t     = m & (T - 1);
                float e0 = c[rg][0][r] + b0;
                float e1 = c[rg][1][r] + b1;
                float e2 = c[rg][2][r] + b2;
                float e3 = c[rg][3][r] + b3;
                size_t base = ((size_t)(b_idx * H + h) * T + t) * HD;
                if (which == 2) {
                    Vo[base + ncol]      = (_Float16)e0;
                    Vo[base + 16 + ncol] = (_Float16)e1;
                    Vo[base + 32 + ncol] = (_Float16)e2;
                    Vo[base + 48 + ncol] = (_Float16)e3;
                } else {
                    const float kfreq = -0.28782313662425572f;   // -ln(10000)/32
                    float ang0 = (float)t * __expf((float)(ncol)      * kfreq);
                    float ang1 = (float)t * __expf((float)(16 + ncol) * kfreq);
                    float s0, c0f, s1, c1f;
                    __sincosf(ang0, &s0, &c0f);
                    __sincosf(ang1, &s1, &c1f);
                    float o0 = e0 * c0f - e2 * s0;
                    float o1 = e1 * c1f - e3 * s1;
                    float o2 = e2 * c0f + e0 * s0;
                    float o3 = e3 * c1f + e1 * s1;
                    _Float16* dst = (which == 0) ? Qo : Ko;
                    dst[base + ncol]      = (_Float16)o0;
                    dst[base + 16 + ncol] = (_Float16)o1;
                    dst[base + 32 + ncol] = (_Float16)o2;
                    dst[base + 48 + ncol] = (_Float16)o3;
                }
            }
        }
      });
}

// ---------------------------------------------------------------------------
// Kernel 2: flash attention per (b,h). 256 thr = 8 waves x 16 queries.
// grid = (T/128, B*H). Output f16 [B,T,D] for the projection GEMM.
// ---------------------------------------------------------------------------
__global__ __launch_bounds__(256) void attn_kernel(
    const _Float16* __restrict__ Q, const _Float16* __restrict__ Kq,
    const _Float16* __restrict__ V, _Float16* __restrict__ O)
{
    __shared__ __align__(16) _Float16 Ks[32 * 64];     // [key][hd]
    __shared__ __align__(16) _Float16 Vt[64 * 32];     // [hd][key]
    __shared__ __align__(16) _Float16 Ps[8][16 * 32];  // per-wave P staging

    const int tid  = threadIdx.x;
    const int lane = tid & 31;
    const int wave = tid >> 5;
    const int bh   = blockIdx.y;
    const int q0   = blockIdx.x * 128 + wave * 16;

    const _Float16* Qb = Q  + (size_t)bh * T * HD;
    const _Float16* Kb = Kq + (size_t)bh * T * HD;
    const _Float16* Vb = V  + (size_t)bh * T * HD;

    v16h qa0 = frag_a_from(Qb + (size_t)q0 * HD,      HD, lane);
    v16h qa1 = frag_a_from(Qb + (size_t)q0 * HD + 32, HD, lane);

    v8f o[4];
#pragma unroll
    for (int i = 0; i < 4; ++i) o[i] = zero8();
    float mrow[8], lrow[8];
#pragma unroll
    for (int r = 0; r < 8; ++r) { mrow[r] = -1e30f; lrow[r] = 0.0f; }

    for (int kt = 0; kt < T; kt += 32) {
        __syncthreads();
        {   // cooperative staging: 256 thr x 8 halves each
            int key = tid >> 3;
            int ch  = (tid & 7) * 8;
            *(v8h*)&Ks[key * 64 + ch] = *(const v8h*)&Kb[(size_t)(kt + key) * HD + ch];
            v8h vv = *(const v8h*)&Vb[(size_t)(kt + key) * HD + ch];
#pragma unroll
            for (int e = 0; e < 8; ++e) Vt[(ch + e) * 32 + key] = vv[e];
        }
        __syncthreads();

        v8f s0 = zero8(), s1 = zero8();
#pragma unroll
        for (int ks = 0; ks < 2; ++ks) {
            v16h a  = ks ? qa1 : qa0;
            v16h b0 = frag_b_from(&Ks[0]       + ks * 32, 64, lane);
            v16h b1 = frag_b_from(&Ks[16 * 64] + ks * 32, 64, lane);
            s0 = WMMA_F16(a, b0, s0);
            s1 = WMMA_F16(a, b1, s1);
        }

        _Float16* Pw = &Ps[wave][0];
        const int ncol = lane & 15;
#pragma unroll
        for (int r = 0; r < 8; ++r) {
            float v0 = s0[r] * 0.125f;
            float v1 = s1[r] * 0.125f;
            float tmax = fmaxf(v0, v1);
#pragma unroll
            for (int mk = 8; mk >= 1; mk >>= 1) tmax = fmaxf(tmax, __shfl_xor(tmax, mk, 32));
            float nm    = fmaxf(mrow[r], tmax);
            float scale = __expf(mrow[r] - nm);
            float p0 = __expf(v0 - nm);
            float p1 = __expf(v1 - nm);
            float ps = p0 + p1;
#pragma unroll
            for (int mk = 8; mk >= 1; mk >>= 1) ps += __shfl_xor(ps, mk, 32);
            lrow[r] = lrow[r] * scale + ps;
            mrow[r] = nm;
#pragma unroll
            for (int f = 0; f < 4; ++f) o[f][r] *= scale;
            int row = r + ((lane & 16) >> 1);
            Pw[row * 32 + ncol]      = (_Float16)p0;
            Pw[row * 32 + 16 + ncol] = (_Float16)p1;
        }
        v16h pa = frag_a_from(Pw, 32, lane);   // same-wave LDS RAW: DS ops in-order
#pragma unroll
        for (int nt = 0; nt < 4; ++nt) {
            v16h b = frag_b_from(&Vt[nt * 16 * 32], 32, lane);
            o[nt]  = WMMA_F16(pa, b, o[nt]);
        }
    }

    const int ncol  = lane & 15;
    const int b_idx = bh >> 4;
    const int h     = bh & 15;
#pragma unroll
    for (int r = 0; r < 8; ++r) {
        int   row = q0 + r + ((lane & 16) >> 1);
        float inv = 1.0f / lrow[r];
        size_t base = ((size_t)b_idx * T + row) * D + h * HD;
        O[base + ncol]      = (_Float16)(o[0][r] * inv);
        O[base + 16 + ncol] = (_Float16)(o[1][r] * inv);
        O[base + 32 + ncol] = (_Float16)(o[2][r] * inv);
        O[base + 48 + ncol] = (_Float16)(o[3][r] * inv);
    }
}

// ---------------------------------------------------------------------------
// Kernel 3: out = attnO @ Wout^T + bout (fp32). grid = (1024/64, 8192/128)
// ---------------------------------------------------------------------------
__global__ __launch_bounds__(128) void proj_kernel(
    const _Float16* __restrict__ A, const _Float16* __restrict__ W,
    const float* __restrict__ bias, float* __restrict__ out)
{
    gemm_core(A, W, Bsz * T, D,
      [=] __device__ (v8f (&c)[2][4], int m0, int n0, int wave, int lane) {
        const int ncol  = lane & 15;
        const int half8 = (lane & 16) >> 1;
#pragma unroll
        for (int rg = 0; rg < 2; ++rg) {
#pragma unroll
            for (int r = 0; r < 8; ++r) {
                int m = m0 + wave * 32 + rg * 16 + r + half8;
                size_t base = (size_t)m * D + n0;
                out[base + ncol]      = c[rg][0][r] + bias[n0 + ncol];
                out[base + 16 + ncol] = c[rg][1][r] + bias[n0 + 16 + ncol];
                out[base + 32 + ncol] = c[rg][2][r] + bias[n0 + 32 + ncol];
                out[base + 48 + ncol] = c[rg][3][r] + bias[n0 + 48 + ncol];
            }
        }
      });
}

extern "C" void kernel_launch(void* const* d_in, const int* in_sizes, int n_in,
                              void* d_out, int out_size, void* d_ws, size_t ws_size,
                              hipStream_t stream) {
    const float* x    = (const float*)d_in[0];
    const float* Wqkv = (const float*)d_in[1];
    const float* bqkv = (const float*)d_in[2];
    const float* Wout = (const float*)d_in[3];
    const float* bout = (const float*)d_in[4];
    float* out = (float*)d_out;

    const size_t NX  = (size_t)Bsz * T * D;   // 8,388,608
    const size_t NWQ = (size_t)3 * D * D;     // 3,145,728
    const size_t NWO = (size_t)D * D;         // 1,048,576

    _Float16* x16  = (_Float16*)d_ws;
    _Float16* wq16 = x16  + NX;
    _Float16* wo16 = wq16 + NWQ;
    _Float16* Qh   = wo16 + NWO;
    _Float16* Kh   = Qh   + NX;
    _Float16* Vh   = Kh   + NX;
    _Float16* AO   = Vh   + NX;

    f32_to_f16<<<dim3((unsigned)(NX  / 256)), 256, 0, stream>>>(x,    x16,  (int)NX);
    f32_to_f16<<<dim3((unsigned)(NWQ / 256)), 256, 0, stream>>>(Wqkv, wq16, (int)NWQ);
    f32_to_f16<<<dim3((unsigned)(NWO / 256)), 256, 0, stream>>>(Wout, wo16, (int)NWO);

    dim3 g1(3 * D / 64, Bsz * T / 128);       // (48,64)
    qkv_rope_kernel<<<g1, 128, 0, stream>>>(x16, wq16, bqkv, Qh, Kh, Vh);

    dim3 g2(T / 128, Bsz * H);                // (16,64)
    attn_kernel<<<g2, 256, 0, stream>>>(Qh, Kh, Vh, AO);

    dim3 g3(D / 64, Bsz * T / 128);           // (16,64)
    proj_kernel<<<g3, 128, 0, stream>>>(AO, wo16, bout, out);
}
